// HeteroGATRefiner_4475355922469
// MI455X (gfx1250) — compile-verified, matrix operand order checked by reading, not run on hardware
//
#include <hip/hip_runtime.h>
#include <hip/hip_bf16.h>
#include <cstdint>
#include <cstddef>

// ---------------------------------------------------------------------------
// HeteroGATRefiner forward for MI455X (gfx1250, wave32, WMMA).
//
// GEMMs computed transposed (D[f][n] = sum_k WT[f][k] * h[n][k]) so that BOTH
// WMMA operand fragments are built from contiguous global_load_b128 per lane:
//   A = WT (weights transposed+bf16-converted once per launch, L2 resident)
//   B = node features, bf16 row-major -> K runs along the contiguous axis.
// Edge softmax/scatter phases are irregular & bandwidth bound: scalar f32 with
// global_atomic_add_f32 and float-max-via-int atomics.
// ---------------------------------------------------------------------------

typedef __attribute__((ext_vector_type(16))) __bf16        v16bf;
typedef __attribute__((ext_vector_type(8)))  float         v8f;
typedef __attribute__((ext_vector_type(4)))  unsigned int  u32x4;

union AFrag { v16bf bf; u32x4 q[2]; };
union CFrag { v8f f; u32x4 q[2]; };

__device__ __forceinline__ unsigned short f32_to_bf16(float f) {
  unsigned int u = __float_as_uint(f);
  u += 0x7FFFu + ((u >> 16) & 1u);          // round-to-nearest-even
  return (unsigned short)(u >> 16);
}
__device__ __forceinline__ float bf16_to_f32(unsigned short s) {
  return __uint_as_float(((unsigned int)s) << 16);
}

// float atomic-max via signed/unsigned int atomics (valid for non-NaN floats)
__device__ __forceinline__ void atomicMaxF(float* addr, float val) {
  if (val >= 0.f) atomicMax((int*)addr, __float_as_int(val));
  else            atomicMin((unsigned int*)addr, __float_as_uint(val));
}

// ---------------------------------------------------------------------------
// Transposed bf16 WMMA GEMM.
//   WT : [Nw, K] bf16 (weights, transposed)      -> WMMA A operand
//   Hb : [M,  K] bf16 (node features, row-major) -> WMMA B operand
//   out[node][feat] = act( (h @ W)[node][feat] + bias[feat] )
// Per-wave tile: 16 features x 64 nodes; K stepped by 32.
// ACT: 0=none, 1=relu.  OBF: 0 -> f32 out, 1 -> bf16 out.
// Grid: x = ceil(M/64) node tiles, y = Nw/128 feature tiles, 256 thr (8 waves).
// ---------------------------------------------------------------------------
template <int ACT, int OBF>
__global__ void gemm_wmma_bf16_t(const unsigned short* __restrict__ WT,
                                 const unsigned short* __restrict__ Hb,
                                 const float* __restrict__ bias,
                                 void* __restrict__ out,
                                 int M, int K, int Nw) {
  const int wave   = threadIdx.x >> 5;
  const int lane   = threadIdx.x & 31;
  const int lanelo = lane & 15;
  const int lanehi = lane >> 4;           // 0 or 1
  const int fBase  = blockIdx.y * 128 + wave * 16;   // feature rows (in range)
  const int nodeBase = blockIdx.x * 64;

  // A-fragment source row: feature fBase+lanelo, K contiguous.
  const unsigned short* aRow = WT + (size_t)(fBase + lanelo) * K;

  // B-fragment node per column group (clamped; stores are guarded).
  int nd[4];
#pragma unroll
  for (int t = 0; t < 4; ++t) {
    int n = nodeBase + t * 16 + lanelo;
    nd[t] = n < M ? n : M - 1;
  }

  v8f acc[4] = {};

  for (int kk = 0; kk < K; kk += 32) {
    // 16-bit fragment layout: lanehi=0 -> elems = K kk+0..7, kk+16..23
    //                         lanehi=1 -> elems = K kk+8..15, kk+24..31
    AFrag a;
    const unsigned short* ap = aRow + kk + lanehi * 8;
    a.q[0] = *reinterpret_cast<const u32x4*>(ap);
    a.q[1] = *reinterpret_cast<const u32x4*>(ap + 16);
#pragma unroll
    for (int t = 0; t < 4; ++t) {
      AFrag b;
      const unsigned short* bp = Hb + (size_t)nd[t] * K + kk + lanehi * 8;
      b.q[0] = *reinterpret_cast<const u32x4*>(bp);
      b.q[1] = *reinterpret_cast<const u32x4*>(bp + 16);
      acc[t] = __builtin_amdgcn_wmma_f32_16x16x32_bf16(
          false, a.bf, false, b.bf, (short)0, acc[t], false, false);
    }
  }

  float bv[8];
#pragma unroll
  for (int v = 0; v < 8; ++v) bv[v] = bias[fBase + lanehi * 8 + v];

#pragma unroll
  for (int t = 0; t < 4; ++t) {
    const int node = nodeBase + t * 16 + lanelo;
    if (node >= M) continue;
    float vals[8];
#pragma unroll
    for (int v = 0; v < 8; ++v) {
      float x = acc[t][v] + bv[v];            // D row = fBase + lanehi*8 + v
      if (ACT == 1) x = fmaxf(x, 0.f);
      vals[v] = x;
    }
    if (OBF) {
      u32x4 pk;
#pragma unroll
      for (int i = 0; i < 4; ++i)
        pk[i] = (unsigned)f32_to_bf16(vals[2 * i]) |
                ((unsigned)f32_to_bf16(vals[2 * i + 1]) << 16);
      *reinterpret_cast<u32x4*>((unsigned short*)out + (size_t)node * Nw +
                                fBase + lanehi * 8) = pk;
    } else {
      CFrag c;
#pragma unroll
      for (int v = 0; v < 8; ++v) c.f[v] = vals[v];
      float* op = (float*)out + (size_t)node * Nw + fBase + lanehi * 8;
      *reinterpret_cast<u32x4*>(op)     = c.q[0];
      *reinterpret_cast<u32x4*>(op + 4) = c.q[1];
    }
  }
}

// ---------------------------------------------------------------------------
// misc small kernels
// ---------------------------------------------------------------------------
__global__ void fill_f32_kernel(float* __restrict__ p, float v, int n) {
  int i = blockIdx.x * blockDim.x + threadIdx.x;
  if (i < n) p[i] = v;
}

// WT[f*K + k] = bf16(W[k*Nw + f])  (transpose + convert; contiguous writes)
__global__ void cvt_transpose_bf16_kernel(const float* __restrict__ W,
                                          unsigned short* __restrict__ WT,
                                          int K, int Nw) {
  int idx = blockIdx.x * blockDim.x + threadIdx.x;
  if (idx >= K * Nw) return;
  int f = idx / K, k = idx - f * K;
  WT[idx] = f32_to_bf16(W[(size_t)k * Nw + f]);
}

__global__ void encoder_kernel(const float* __restrict__ x,
                               const float* __restrict__ W,
                               const float* __restrict__ b,
                               unsigned short* __restrict__ hbf, int total) {
  int idx = blockIdx.x * blockDim.x + threadIdx.x;
  if (idx >= total) return;
  int n = idx >> 7, j = idx & 127;
  const float* xr = x + (size_t)n * 4;
  float v = b[j];
#pragma unroll
  for (int k = 0; k < 4; ++k) v += xr[k] * W[k * 128 + j];
  hbf[idx] = f32_to_bf16(v);
}

// ---------------------------------------------------------------------------
// edge phases: one thread per (edge, head)
// ---------------------------------------------------------------------------
__global__ void edge_score_kernel(const float* __restrict__ xl,
                                  const float* __restrict__ xr,
                                  const int* __restrict__ src,
                                  const int* __restrict__ dst,
                                  const float* __restrict__ eattr,  // may be null
                                  const float* __restrict__ We,     // may be null
                                  const float* __restrict__ att,
                                  float* __restrict__ score,
                                  float* __restrict__ smax, int total) {
  int idx = blockIdx.x * blockDim.x + threadIdx.x;
  if (idx >= total) return;
  int e = idx >> 3, h = idx & 7;
  int s = src[e], d = dst[e];
  const float ea = We ? eattr[e] : 0.f;
  const float* pl = xl + (size_t)s * 128 + h * 16;
  const float* pr = xr + (size_t)d * 128 + h * 16;
  const float* pa = att + h * 16;
  float sc = 0.f;
#pragma unroll
  for (int c = 0; c < 16; ++c) {
    float v = pl[c] + pr[c];
    if (We) v += ea * We[h * 16 + c];
    v = v > 0.f ? v : 0.2f * v;       // LeakyReLU(0.2)
    sc += v * pa[c];
  }
  score[idx] = sc;
  atomicMaxF(&smax[d * 8 + h], sc);
}

__global__ void edge_exp_kernel(float* __restrict__ score,
                                const int* __restrict__ dst,
                                const float* __restrict__ smax,
                                float* __restrict__ denom, int total) {
  int idx = blockIdx.x * blockDim.x + threadIdx.x;
  if (idx >= total) return;
  int e = idx >> 3, h = idx & 7;
  int d = dst[e];
  float ex = __expf(score[idx] - smax[d * 8 + h]);
  score[idx] = ex;
  atomicAdd(&denom[d * 8 + h], ex);
}

__global__ void edge_scatter_kernel(const float* __restrict__ score,
                                    const float* __restrict__ xl,
                                    const int* __restrict__ src,
                                    const int* __restrict__ dst,
                                    const float* __restrict__ denom,
                                    float* __restrict__ outsum, int total) {
  int idx = blockIdx.x * blockDim.x + threadIdx.x;
  if (idx >= total) return;
  int e = idx >> 3, h = idx & 7;
  int s = src[e], d = dst[e];
  float alpha = score[idx] / (denom[d * 8 + h] + 1e-16f);
  const float* pl = xl + (size_t)s * 128 + h * 16;
  float* po = outsum + (size_t)d * 128 + h * 16;
#pragma unroll
  for (int c = 0; c < 16; ++c) atomicAdd(&po[c], alpha * pl[c]);
}

__global__ void combine_elu_kernel(const float* __restrict__ outsum,
                                   const float* __restrict__ boP,
                                   const float* __restrict__ boA,
                                   const float* __restrict__ boL,
                                   unsigned short* __restrict__ hbf, int total) {
  int idx = blockIdx.x * blockDim.x + threadIdx.x;
  if (idx >= total) return;
  int j = idx & 127;
  float v = outsum[idx] + boP[j] + boA[j] + boL[j];
  float hv = v > 0.f ? v : (__expf(v) - 1.f);   // ELU
  hbf[idx] = f32_to_bf16(hv);
}

__global__ void mean_kernel(const unsigned short* __restrict__ hbf,
                            float* __restrict__ g, int Nn) {
  int tid = threadIdx.x;            // 256
  int j = tid & 127;
  float s = 0.f;
  for (int n = blockIdx.x * 2 + (tid >> 7); n < Nn; n += gridDim.x * 2)
    s += bf16_to_f32(hbf[(size_t)n * 128 + j]);
  atomicAdd(&g[j], s);
}

__global__ void critic_kernel(const float* __restrict__ g,
                              const float* __restrict__ W1,
                              const float* __restrict__ b1,
                              const float* __restrict__ W2,
                              const float* __restrict__ b2,
                              float* __restrict__ out_value, int Nn) {
  __shared__ float gs[128];
  __shared__ float red[128];
  int tid = threadIdx.x;            // 128
  gs[tid] = g[tid] * (1.f / (float)Nn);
  __syncthreads();
  float a = b1[tid];
  for (int k = 0; k < 128; ++k) a += gs[k] * W1[k * 128 + tid];
  float hid = fmaxf(a, 0.f);
  red[tid] = hid * W2[tid];
  __syncthreads();
  for (int off = 64; off > 0; off >>= 1) {
    if (tid < off) red[tid] += red[tid + off];
    __syncthreads();
  }
  if (tid == 0) out_value[0] = red[0] + b2[0];
}

// ---------------------------------------------------------------------------
// host
// ---------------------------------------------------------------------------
extern "C" void kernel_launch(void* const* d_in, const int* in_sizes, int n_in,
                              void* d_out, int out_size, void* d_ws, size_t ws_size,
                              hipStream_t stream) {
  (void)n_in; (void)out_size; (void)ws_size;
  const int Nn = in_sizes[0] / 4;   // 50000

  auto F = [&](int i) { return (const float*)d_in[i]; };
  auto I = [&](int i) { return (const int*)d_in[i]; };

  // d_in index map (recursive insertion-order flatten of setup_inputs()):
  // 0:x 1:enc.W 2:enc.b
  // layer l base = 3 + l*20:
  //   phys:  +0 Wl +1 bl +2 Wr +3 br +4 att +5 bo +6 We
  //   align: +7 Wl +8 bl +9 Wr +10 br +11 att +12 bo
  //   logic: +13 Wl +14 bl +15 Wr +16 br +17 att +18 bo +19 We
  const int ACTOR = 43, CRITIC = 47, EDGE0 = 51;

  // ---- workspace bump allocator ----
  uintptr_t base = (uintptr_t)d_ws;
  auto alloc = [&](size_t bytes) -> void* {
    uintptr_t p = (base + 255) & ~(uintptr_t)255;
    base = p + bytes;
    return (void*)p;
  };
  const int E0 = in_sizes[EDGE0], E1 = in_sizes[EDGE0 + 3], E2 = in_sizes[EDGE0 + 5];
  int Emax = E0 > E1 ? E0 : E1; if (E2 > Emax) Emax = E2;

  unsigned short* hbf    = (unsigned short*)alloc((size_t)Nn * 128 * 2);
  unsigned short* hidbf  = (unsigned short*)alloc((size_t)Nn * 128 * 2);
  float* xl     = (float*)alloc((size_t)Nn * 128 * 4);
  float* xr     = (float*)alloc((size_t)Nn * 128 * 4);
  float* outsum = (float*)alloc((size_t)Nn * 128 * 4);
  float* score  = (float*)alloc((size_t)Emax * 8 * 4);
  float* smax   = (float*)alloc((size_t)Nn * 8 * 4);
  float* denom  = (float*)alloc((size_t)Nn * 8 * 4);
  float* gmean  = (float*)alloc(128 * 4);
  unsigned short* wbf = (unsigned short*)alloc((size_t)(13 * 16384 + 128 * 512) * 2);

  // ---- transpose + convert weights to bf16 [Nw,K] ----
  unsigned short* wlT[2][3];
  unsigned short* wrT[2][3];
  int slot = 0;
  for (int l = 0; l < 2; ++l) {
    const int relOff[3] = {0, 7, 13};
    for (int r = 0; r < 3; ++r) {
      int b0 = 3 + l * 20 + relOff[r];
      wlT[l][r] = wbf + (size_t)slot++ * 16384;
      cvt_transpose_bf16_kernel<<<64, 256, 0, stream>>>(F(b0 + 0), wlT[l][r], 128, 128);
      wrT[l][r] = wbf + (size_t)slot++ * 16384;
      cvt_transpose_bf16_kernel<<<64, 256, 0, stream>>>(F(b0 + 2), wrT[l][r], 128, 128);
    }
  }
  unsigned short* aW1T = wbf + (size_t)slot++ * 16384;
  cvt_transpose_bf16_kernel<<<64, 256, 0, stream>>>(F(ACTOR + 0), aW1T, 128, 128);
  unsigned short* aW2T = wbf + (size_t)slot * 16384;   // [512,128]
  cvt_transpose_bf16_kernel<<<256, 256, 0, stream>>>(F(ACTOR + 2), aW2T, 128, 512);

  const int totNH = Nn * 128;
  const int gNH = (totNH + 255) / 256;

  // ---- encoder ----
  encoder_kernel<<<gNH, 256, 0, stream>>>(F(0), F(1), F(2), hbf, totNH);

  auto gemm = [&](const unsigned short* WT, const unsigned short* Hb,
                  const float* bias, void* out, int M, int K, int Nw,
                  int act, int obf) {
    dim3 gb((M + 63) / 64, Nw / 128);
    if (act == 1 && obf == 1)
      gemm_wmma_bf16_t<1, 1><<<gb, 256, 0, stream>>>(WT, Hb, bias, out, M, K, Nw);
    else
      gemm_wmma_bf16_t<0, 0><<<gb, 256, 0, stream>>>(WT, Hb, bias, out, M, K, Nw);
  };

  // ---- GAT layers ----
  for (int l = 0; l < 2; ++l) {
    fill_f32_kernel<<<gNH, 256, 0, stream>>>(outsum, 0.f, totNH);

    struct RelDesc { int b0; bool hasWe; const int* src; const int* dst; const float* attr; int E; };
    RelDesc rels[3] = {
      {3 + l * 20 + 0,  true,  I(EDGE0 + 0), I(EDGE0 + 1), F(EDGE0 + 2), E0},
      {3 + l * 20 + 7,  false, I(EDGE0 + 3), I(EDGE0 + 4), nullptr,      E1},
      {3 + l * 20 + 13, true,  I(EDGE0 + 5), I(EDGE0 + 6), F(EDGE0 + 7), E2},
    };

    for (int r = 0; r < 3; ++r) {
      const RelDesc& rd = rels[r];
      gemm(wlT[l][r], hbf, F(rd.b0 + 1), xl, Nn, 128, 128, 0, 0);
      gemm(wrT[l][r], hbf, F(rd.b0 + 3), xr, Nn, 128, 128, 0, 0);

      const int totNH8 = Nn * 8, gN8 = (totNH8 + 255) / 256;
      fill_f32_kernel<<<gN8, 256, 0, stream>>>(smax, -__builtin_inff(), totNH8);
      fill_f32_kernel<<<gN8, 256, 0, stream>>>(denom, 0.f, totNH8);

      const int totE = rd.E * 8, gE = (totE + 255) / 256;
      edge_score_kernel<<<gE, 256, 0, stream>>>(
          xl, xr, rd.src, rd.dst, rd.attr,
          rd.hasWe ? F(rd.b0 + 6) : nullptr, F(rd.b0 + 4),
          score, smax, totE);
      edge_exp_kernel<<<gE, 256, 0, stream>>>(score, rd.dst, smax, denom, totE);
      edge_scatter_kernel<<<gE, 256, 0, stream>>>(score, xl, rd.src, rd.dst,
                                                  denom, outsum, totE);
    }
    combine_elu_kernel<<<gNH, 256, 0, stream>>>(
        outsum, F(3 + l * 20 + 5), F(3 + l * 20 + 12), F(3 + l * 20 + 18),
        hbf, totNH);
  }

  // ---- actor head: relu(h@W1+b1) @ W2 + b2 -> logits ----
  gemm(aW1T, hbf, F(ACTOR + 1), hidbf, Nn, 128, 128, 1, 1);
  gemm(aW2T, hidbf, F(ACTOR + 3), (float*)d_out, Nn, 128, 512, 0, 0);

  // ---- critic head on graph mean ----
  fill_f32_kernel<<<1, 128, 0, stream>>>(gmean, 0.f, 128);
  mean_kernel<<<256, 256, 0, stream>>>(hbf, gmean, Nn);
  critic_kernel<<<1, 128, 0, stream>>>(gmean, F(CRITIC + 0), F(CRITIC + 1),
                                       F(CRITIC + 2), F(CRITIC + 3),
                                       (float*)d_out + (size_t)Nn * 512, Nn);
}
